// GNNEncoder_41601053229566
// MI455X (gfx1250) — matrix operand order repaired and checked by reading, not compiled, not run on hardware
//
#include <hip/hip_runtime.h>

#define D 128
#define NGR 64

typedef __bf16 v16bf __attribute__((ext_vector_type(16)));
typedef float  v8f   __attribute__((ext_vector_type(8)));

__device__ __forceinline__ __bf16 f2bf(float f) {
  unsigned u = __builtin_bit_cast(unsigned, f);
  u += 0x7FFFu + ((u >> 16) & 1u);               // round-to-nearest-even
  unsigned short h = (unsigned short)(u >> 16);
  return __builtin_bit_cast(__bf16, h);
}

// ---------------------------------------------------------------------------
// Pack a 128x128 row-major f32 weight matrix into WMMA B-fragment order:
// frag f = kb*8+nb, lane l: 16 contiguous bf16 = W[kb*32 + (l>>4)*8 + (i&7) + (i>>3)*16][nb*16 + (l&15)]
// ---------------------------------------------------------------------------
__global__ void pack_w_bf16(const float* __restrict__ W, __bf16* __restrict__ Bp) {
  int t = blockIdx.x * blockDim.x + threadIdx.x;
  if (t >= 1024) return;
  int lane = t & 31, f = t >> 5;
  int kb = f >> 3, nb = f & 7;
  int g = lane >> 4, col = lane & 15;
  __bf16* o = Bp + (size_t)t * 16;
#pragma unroll
  for (int i = 0; i < 16; ++i)
    o[i] = f2bf(W[(size_t)(kb * 32 + g * 8 + (i & 7) + (i >> 3) * 16) * D + nb * 16 + col]);
}

// ---------------------------------------------------------------------------
// C[M,128] = act(A[M,128]) @ W (pre-packed bf16 frags) [+ obias], f32 accum.
// act(v) = abias ? relu(v + abias[col]) : v    (fuses GCN bias+relu into the
// next layer's A-fragment conversion -> saves a full 2x51MB memory pass).
// 256 threads = 8 waves, wave owns a 16x128 strip; 32 v_wmma per wave.
// ---------------------------------------------------------------------------
__global__ void __launch_bounds__(256) gemm_wmma_bf16(
    const float* __restrict__ A, const __bf16* __restrict__ Bp,
    const float* __restrict__ abias, const float* __restrict__ obias,
    float* __restrict__ C, int M)
{
  const int wave = threadIdx.x >> 5;
  const int lane = threadIdx.x & 31;
  const int g    = lane >> 4;                    // 0/1 lane half
  const int m0   = (blockIdx.x * 8 + wave) * 16;
  int row = m0 + (lane & 15);
  int rc  = row < M ? row : (M - 1);
  const float* Ar = A + (size_t)rc * D;

  // A fragments, ISA 16-bit 16x32 layout:
  //  lanes 0-15 : V0..3 = K 0..7 ; V4..7 = K16..23   (g=0)
  //  lanes16-31 : V0..3 = K 8..15; V4..7 = K24..31   (g=1)
  v16bf a[4];
#pragma unroll
  for (int kb = 0; kb < 4; ++kb) {
    const int k0 = kb * 32 + g * 8;
    const int k1 = kb * 32 + 16 + g * 8;
    const float4* p0 = reinterpret_cast<const float4*>(Ar + k0);
    const float4* p1 = reinterpret_cast<const float4*>(Ar + k1);
    float4 f0 = p0[0], f1 = p0[1], f2 = p1[0], f3 = p1[1];
    if (abias) {                                  // fused bias + relu on input
      const float4* b0 = reinterpret_cast<const float4*>(abias + k0);
      const float4* b1 = reinterpret_cast<const float4*>(abias + k1);
      float4 c0 = b0[0], c1 = b0[1], c2 = b1[0], c3 = b1[1];
      f0.x = fmaxf(f0.x + c0.x, 0.f); f0.y = fmaxf(f0.y + c0.y, 0.f);
      f0.z = fmaxf(f0.z + c0.z, 0.f); f0.w = fmaxf(f0.w + c0.w, 0.f);
      f1.x = fmaxf(f1.x + c1.x, 0.f); f1.y = fmaxf(f1.y + c1.y, 0.f);
      f1.z = fmaxf(f1.z + c1.z, 0.f); f1.w = fmaxf(f1.w + c1.w, 0.f);
      f2.x = fmaxf(f2.x + c2.x, 0.f); f2.y = fmaxf(f2.y + c2.y, 0.f);
      f2.z = fmaxf(f2.z + c2.z, 0.f); f2.w = fmaxf(f2.w + c2.w, 0.f);
      f3.x = fmaxf(f3.x + c3.x, 0.f); f3.y = fmaxf(f3.y + c3.y, 0.f);
      f3.z = fmaxf(f3.z + c3.z, 0.f); f3.w = fmaxf(f3.w + c3.w, 0.f);
    }
    v16bf av;
    av[0]  = f2bf(f0.x); av[1]  = f2bf(f0.y); av[2]  = f2bf(f0.z); av[3]  = f2bf(f0.w);
    av[4]  = f2bf(f1.x); av[5]  = f2bf(f1.y); av[6]  = f2bf(f1.z); av[7]  = f2bf(f1.w);
    av[8]  = f2bf(f2.x); av[9]  = f2bf(f2.y); av[10] = f2bf(f2.z); av[11] = f2bf(f2.w);
    av[12] = f2bf(f3.x); av[13] = f2bf(f3.y); av[14] = f2bf(f3.z); av[15] = f2bf(f3.w);
    a[kb] = av;
  }

  const int crow0   = m0 + g * 8;                // C layout: VGPR r -> M = m0 + g*8 + r
  const int ccol_l  = lane & 15;
  const bool full   = (m0 + 16) <= M;            // wave-uniform fast path

#pragma unroll
  for (int nb = 0; nb < 8; ++nb) {
    v8f acc = {};
#pragma unroll
    for (int kb = 0; kb < 4; ++kb) {
      v16bf bv = *reinterpret_cast<const v16bf*>(
          Bp + ((size_t)((kb << 3) + nb) * 32 + lane) * 16);
      acc = __builtin_amdgcn_wmma_f32_16x16x32_bf16(
          false, a[kb], false, bv, (short)0, acc, false, false);
    }
    const int ccol = nb * 16 + ccol_l;
    const float bb = obias ? obias[ccol] : 0.0f;
    float* Cp = C + (size_t)crow0 * D + ccol;
    if (full) {                                  // branch-free stores
#pragma unroll
      for (int r = 0; r < 8; ++r) Cp[(size_t)r * D] = acc[r] + bb;
    } else {
#pragma unroll
      for (int r = 0; r < 8; ++r)
        if (crow0 + r < M) Cp[(size_t)r * D] = acc[r] + bb;
    }
  }
}

// ---------------------------------------------------------------------------
// Degree / norm
// ---------------------------------------------------------------------------
__global__ void deg_init(float* __restrict__ deg, int n) {
  int i = blockIdx.x * blockDim.x + threadIdx.x;
  if (i < n) deg[i] = 1.0f;                      // self-loop
}
__global__ void deg_acc(const int* __restrict__ dst, float* __restrict__ deg, int E) {
  int e = blockIdx.x * blockDim.x + threadIdx.x;
  if (e < E) atomicAdd(&deg[dst[e]], 1.0f);
}
__global__ void dinv_inplace(float* __restrict__ deg, int n) {
  int i = blockIdx.x * blockDim.x + threadIdx.x;
  if (i < n) deg[i] = rsqrtf(deg[i]);            // deg >= 1 always
}

// h[i] = t[i] * dinv[i]^2   (self-loop contribution, initializes accumulator)
__global__ void self_loop_init(const float* __restrict__ t, const float* __restrict__ dinv,
                               float* __restrict__ h, int n) {
  int tid = blockIdx.x * blockDim.x + threadIdx.x;
  if (tid >= n * (D / 4)) return;
  int node = tid >> 5;
  float s = dinv[node]; s = s * s;
  float4 v = reinterpret_cast<const float4*>(t)[tid];
  float4 o; o.x = v.x * s; o.y = v.y * s; o.z = v.z * s; o.w = v.w * s;
  reinterpret_cast<float4*>(h)[tid] = o;
}

// one wave per edge: h[dst] += t[src] * dinv[src]*dinv[dst]
__global__ void __launch_bounds__(256) scatter_edges(
    const float* __restrict__ t, const int* __restrict__ src, const int* __restrict__ dst,
    const float* __restrict__ dinv, float* __restrict__ h, int E)
{
  int e = blockIdx.x * 8 + (threadIdx.x >> 5);
  int lane = threadIdx.x & 31;
  if (e >= E) return;
  int s = src[e], d = dst[e];
  float w = dinv[s] * dinv[d];
  float4 v = *reinterpret_cast<const float4*>(t + (size_t)s * D + lane * 4);
  float* p = h + (size_t)d * D + lane * 4;
  atomicAdd(p + 0, v.x * w);
  atomicAdd(p + 1, v.y * w);
  atomicAdd(p + 2, v.z * w);
  atomicAdd(p + 3, v.w * w);
}

// ---------------------------------------------------------------------------
// Mean pool with fused bias+relu: sums[g] += relu(h[node]+b), LDS staging.
// ---------------------------------------------------------------------------
__global__ void zero_f32(float* __restrict__ p, int n) {
  int i = blockIdx.x * blockDim.x + threadIdx.x;
  if (i < n) p[i] = 0.0f;
}

__global__ void __launch_bounds__(256) pool_kernel(
    const float* __restrict__ h, const float* __restrict__ bias,
    const int* __restrict__ batch,
    float* __restrict__ sums, float* __restrict__ cnts, int n)
{
  __shared__ float ls[NGR * D];
  __shared__ float lc[NGR];
  for (int i = threadIdx.x; i < NGR * D; i += 256) ls[i] = 0.0f;
  if (threadIdx.x < NGR) lc[threadIdx.x] = 0.0f;
  __syncthreads();
  int per = (n + gridDim.x - 1) / gridDim.x;
  int n0 = blockIdx.x * per;
  int n1 = n0 + per; if (n1 > n) n1 = n;
  int grp = threadIdx.x >> 7;                    // 2 groups of 128 threads
  int c   = threadIdx.x & 127;
  float bc = bias[c];
  for (int node = n0 + grp; node < n1; node += 2) {
    int gid = batch[node];
    float v = fmaxf(h[(size_t)node * D + c] + bc, 0.0f);
    atomicAdd(&ls[gid * D + c], v);
    if (c == 0) atomicAdd(&lc[gid], 1.0f);
  }
  __syncthreads();
  for (int i = threadIdx.x; i < NGR * D; i += 256) atomicAdd(&sums[i], ls[i]);
  if (threadIdx.x < NGR) atomicAdd(&cnts[threadIdx.x], lc[threadIdx.x]);
}

__global__ void mean_kernel(const float* __restrict__ sums, const float* __restrict__ cnts,
                            float* __restrict__ pm) {
  int i = blockIdx.x * blockDim.x + threadIdx.x;
  if (i < NGR * D) pm[i] = sums[i] / fmaxf(cnts[i >> 7], 1.0f);
}

// ---------------------------------------------------------------------------
extern "C" void kernel_launch(void* const* d_in, const int* in_sizes, int n_in,
                              void* d_out, int out_size, void* d_ws, size_t ws_size,
                              hipStream_t stream) {
  (void)n_in; (void)out_size; (void)ws_size;
  const float* x    = (const float*)d_in[0];
  const int*   ei   = (const int*)d_in[1];
  const int*   batch= (const int*)d_in[2];
  const float* W1   = (const float*)d_in[3];
  const float* b1   = (const float*)d_in[4];
  const float* W2   = (const float*)d_in[5];
  const float* b2   = (const float*)d_in[6];
  const float* Wfc  = (const float*)d_in[7];
  const float* bfc  = (const float*)d_in[8];
  float* out = (float*)d_out;

  const int n = in_sizes[0] / D;
  const int E = in_sizes[1] / 2;
  const int* src = ei;
  const int* dst = ei + E;

  char* w = (char*)d_ws;
  auto alloc = [&](size_t bytes) -> char* {
    char* p = w; w += (bytes + 511) & ~(size_t)511; return p;
  };
  __bf16* wp1  = (__bf16*)alloc(1024 * 16 * sizeof(__bf16));
  __bf16* wp2  = (__bf16*)alloc(1024 * 16 * sizeof(__bf16));
  __bf16* wpfc = (__bf16*)alloc(1024 * 16 * sizeof(__bf16));
  float* deg  = (float*)alloc((size_t)n * sizeof(float));       // becomes dinv
  float* t    = (float*)alloc((size_t)n * D * sizeof(float));   // GEMM output
  float* h    = (float*)alloc((size_t)n * D * sizeof(float));   // aggregated features
  float* sums = (float*)alloc((NGR * D + NGR) * sizeof(float));
  float* cnts = sums + NGR * D;
  float* pm   = (float*)alloc(NGR * D * sizeof(float));

  // weight packing + norm
  pack_w_bf16<<<4, 256, 0, stream>>>(W1, wp1);
  pack_w_bf16<<<4, 256, 0, stream>>>(W2, wp2);
  pack_w_bf16<<<4, 256, 0, stream>>>(Wfc, wpfc);
  deg_init<<<(n + 255) / 256, 256, 0, stream>>>(deg, n);
  deg_acc<<<(E + 255) / 256, 256, 0, stream>>>(dst, deg, E);
  dinv_inplace<<<(n + 255) / 256, 256, 0, stream>>>(deg, n);

  const int gblocks = (n + 127) / 128;
  const int qblocks = (n * (D / 4) + 255) / 256;
  const int eblocks = (E + 7) / 8;

  // layer 1: t = x@W1 ; h = norm-aggregate(t)      (bias+relu deferred)
  gemm_wmma_bf16<<<gblocks, 256, 0, stream>>>(x, wp1, nullptr, nullptr, t, n);
  self_loop_init<<<qblocks, 256, 0, stream>>>(t, deg, h, n);
  scatter_edges<<<eblocks, 256, 0, stream>>>(t, src, dst, deg, h, E);

  // layer 2: t = relu(h+b1)@W2 ; h = norm-aggregate(t)   (b2+relu deferred)
  gemm_wmma_bf16<<<gblocks, 256, 0, stream>>>(h, wp2, b1, nullptr, t, n);
  self_loop_init<<<qblocks, 256, 0, stream>>>(t, deg, h, n);
  scatter_edges<<<eblocks, 256, 0, stream>>>(t, src, dst, deg, h, E);

  // mean pool (fused relu(h+b2)) + FC
  zero_f32<<<(NGR * D + NGR + 255) / 256, 256, 0, stream>>>(sums, NGR * D + NGR);
  pool_kernel<<<256, 256, 0, stream>>>(h, b2, batch, sums, cnts, n);
  mean_kernel<<<(NGR * D + 255) / 256, 256, 0, stream>>>(sums, cnts, pm);
  gemm_wmma_bf16<<<1, 256, 0, stream>>>(pm, wpfc, nullptr, bfc, out, NGR);
}
